// AgentMatchingDecoder_13615046328615
// MI455X (gfx1250) — matrix operand order repaired and checked by reading, not compiled, not run on hardware
//
#include <hip/hip_runtime.h>

typedef __attribute__((ext_vector_type(16))) _Float16 v16h;
typedef __attribute__((ext_vector_type(8)))  _Float16 v8h;
typedef __attribute__((ext_vector_type(8)))  float    v8f;

#define BSZ  8
#define NAG  8
#define CDIM 512
#define HWSZ 1024
#define DFFD 2048
#define IMR  128

__device__ __forceinline__ int imin(int a, int b) { return a < b ? a : b; }
__device__ __forceinline__ int imax(int a, int b) { return a > b ? a : b; }

__device__ __forceinline__ v16h cat8(v8h lo, v8h hi) {
    return __builtin_shufflevector(lo, hi, 0,1,2,3,4,5,6,7,8,9,10,11,12,13,14,15);
}

// ---------------------------------------------------------------------------
// f16 fragment loaders (wave32); A and B share this file's contiguous k order.
// A 16x32: lane m=lane&15, K(i) = (lane/16)*16 + i
// B 32x16: lane n=lane&15, K(i) = (lane/16)*16 + i
// C/D f32: M = r + 8*(lane/16), N = lane&15.
// GM/GN guards are address CLAMPS only; K guards zero-fill branchlessly.
// ---------------------------------------------------------------------------

template<bool GM>
__device__ __forceinline__ v16h la_row(const _Float16* __restrict__ A, int lda,
                                       int M, int tm, int kb, int lane) {
    int m = tm * 16 + (lane & 15);
    if (GM) m = imin(m, M - 1);
    const _Float16* p = A + (size_t)m * lda + kb + ((lane >> 4) << 4);
    v8h lo = *(const v8h*)p;
    v8h hi = *(const v8h*)(p + 8);
    return cat8(lo, hi);
}

template<bool GN>
__device__ __forceinline__ v16h lb_wt(const _Float16* __restrict__ B, int ldb,
                                      int N, int tn, int kb, int lane) {
    int n = tn * 16 + (lane & 15);
    if (GN) n = imin(n, N - 1);
    const _Float16* p = B + (size_t)n * ldb + kb + ((lane >> 4) << 4);
    v8h lo = *(const v8h*)p;
    v8h hi = *(const v8h*)(p + 8);
    return cat8(lo, hi);
}

__device__ __forceinline__ v16h la_col_gk(const _Float16* __restrict__ A, int lda,
                                          int K, int tm, int kb, int lane) {
    int m = tm * 16 + (lane & 15);
    int g = lane >> 4;
    v16h a;
#pragma unroll
    for (int i = 0; i < 16; ++i) {
        int k = kb + (g << 4) + i;
        int kc = imin(k, K - 1);
        _Float16 v = A[(size_t)kc * lda + m];
        a[i] = (k < K) ? v : (_Float16)0.0f;
    }
    return a;
}

__device__ __forceinline__ v16h lb_wt_gk(const _Float16* __restrict__ B, int ldb,
                                         int K, int tn, int kb, int lane) {
    int n = tn * 16 + (lane & 15);
    int g = lane >> 4;
    v16h b;
#pragma unroll
    for (int i = 0; i < 16; ++i) {
        int k = kb + (g << 4) + i;
        int kc = imin(k, K - 1);
        _Float16 v = B[(size_t)n * ldb + kc];
        b[i] = (k < K) ? v : (_Float16)0.0f;
    }
    return b;
}

// ---------------------------------------------------------------------------
// Register-blocked wave-tile GEMM: each wave computes an (MB*16) x (NB*16)
// block -> MB*NB WMMAs per k-step from MB+NB fragment loads.
// ALAY: 0 = A row-major fast, 1 = A col-major with K zero-guard
// BLAY: 0 = "Wt" form fast (B[k][n]=src[n*ldb+k]), 2 = "Wt" with K zero-guard
// OUT:  0 = f32, 1 = f16, 2 = both, 3 = f16 transposed (Ch[n*ldct+m])
// grid: blockIdx.y = M block, blockIdx.x*8+wave = N block (no divisions).
// Blocked (MB/NB>1) call sites have M % (16*MB) == 0 and N % (16*NB) == 0.
// ---------------------------------------------------------------------------
template<int ALAY, int BLAY, bool GM, bool GN, bool RELU, bool HASB, int OUT,
         int MB, int NB>
__global__ void gemm_k(const _Float16* __restrict__ A, int lda,
                       const _Float16* __restrict__ B, int ldb,
                       const float* __restrict__ bias,
                       float* __restrict__ Cf, _Float16* __restrict__ Ch,
                       int ldc, int ldct,
                       int M, int N, int K, float alpha) {
    int tilesN = (N + 15) >> 4;
    int grp = blockIdx.x * (blockDim.x >> 5) + (threadIdx.x >> 5);
    int tn0 = grp * NB;
    if (tn0 >= tilesN) return;   // wave-uniform: EXEC all-1s at WMMA
    int tm0 = blockIdx.y * MB;
    int lane = threadIdx.x & 31;

    v8f zero = {};
    v8f acc[MB][NB];
#pragma unroll
    for (int im = 0; im < MB; ++im)
#pragma unroll
        for (int in = 0; in < NB; ++in) acc[im][in] = zero;

    for (int kb = 0; kb < K; kb += 32) {
        v16h a[MB], b[NB];
#pragma unroll
        for (int im = 0; im < MB; ++im) {
            if constexpr (ALAY == 0) a[im] = la_row<GM>(A, lda, M, tm0 + im, kb, lane);
            else                     a[im] = la_col_gk(A, lda, K, tm0 + im, kb, lane);
        }
#pragma unroll
        for (int in = 0; in < NB; ++in) {
            if constexpr (BLAY == 0) b[in] = lb_wt<GN>(B, ldb, N, tn0 + in, kb, lane);
            else                     b[in] = lb_wt_gk(B, ldb, K, tn0 + in, kb, lane);
        }
#pragma unroll
        for (int im = 0; im < MB; ++im)
#pragma unroll
            for (int in = 0; in < NB; ++in)
                acc[im][in] = __builtin_amdgcn_wmma_f32_16x16x32_f16(
                    false, a[im], false, b[in], (short)0, acc[im][in], false, false);
    }

    int g = lane >> 4;
#pragma unroll
    for (int im = 0; im < MB; ++im) {
#pragma unroll
        for (int in = 0; in < NB; ++in) {
            int n = (tn0 + in) * 16 + (lane & 15);
#pragma unroll
            for (int r = 0; r < 8; ++r) {
                int m = (tm0 + im) * 16 + r + (g << 3);
                bool ok = (!GM || m < M) && (!GN || n < N);
                if (ok) {
                    float v = acc[im][in][r] * alpha;
                    if (HASB) v += bias[n];
                    if (RELU) v = fmaxf(v, 0.0f);
                    if constexpr (OUT == 0) {
                        Cf[(size_t)m * ldc + n] = v;
                    } else if constexpr (OUT == 1) {
                        Ch[(size_t)m * ldc + n] = (_Float16)v;
                    } else if constexpr (OUT == 2) {
                        Cf[(size_t)m * ldc + n] = v;
                        Ch[(size_t)m * ldc + n] = (_Float16)v;
                    } else {
                        Ch[(size_t)n * ldct + m] = (_Float16)v;
                    }
                }
            }
        }
    }
}

// ---------------------------------------------------------------------------
// fp32 -> f16 conversion
// ---------------------------------------------------------------------------
__global__ void cvt_h_k(const float* __restrict__ s, _Float16* __restrict__ d, int n) {
    int i = blockIdx.x * blockDim.x + threadIdx.x;
    if (i < n) d[i] = (_Float16)s[i];
}

// ---------------------------------------------------------------------------
// argmax over agent axis (first-occurrence, like jnp.argmax)
// ---------------------------------------------------------------------------
__global__ void argmax_k(const float* __restrict__ Sas, const float* __restrict__ Sqa,
                         int* __restrict__ a_idx, int* __restrict__ q_idx) {
    int p = blockIdx.x * blockDim.x + threadIdx.x;
    if (p >= HWSZ) return;
    float bv = Sas[p]; int bi = 0;
    for (int a = 1; a < NAG; ++a) {
        float v = Sas[(size_t)a * HWSZ + p];
        if (v > bv) { bv = v; bi = a; }
    }
    a_idx[p] = bi;
    bv = Sqa[(size_t)p * NAG]; bi = 0;
    for (int a = 1; a < NAG; ++a) {
        float v = Sqa[(size_t)p * NAG + a];
        if (v > bv) { bv = v; bi = a; }
    }
    q_idx[p] = bi;
}

// ---------------------------------------------------------------------------
// Softmax over j of L[h, j] + alignment bias; writes TRANSPOSED f16 output:
// Pth[j, h] = softmax_j(L[h,:] + bias)[j]
// ---------------------------------------------------------------------------
__global__ void softmax_t_k(const float* __restrict__ L, float* __restrict__ Ptf,
                            _Float16* __restrict__ Pth,
                            const int* __restrict__ a_idx, const int* __restrict__ q_idx) {
    __shared__ float red[8];
    int h = blockIdx.x;
    int tid = threadIdx.x, lane = tid & 31, wid = tid >> 5;
    int ai = a_idx[h];
    const float* row = L + (size_t)h * HWSZ;

    float mx = -3.4e38f;
    for (int j = tid; j < HWSZ; j += blockDim.x) {
        float v = row[j] + ((q_idx[j] == ai) ? 0.0f : -1.0e6f);
        mx = fmaxf(mx, v);
    }
    for (int off = 16; off; off >>= 1) mx = fmaxf(mx, __shfl_xor(mx, off, 32));
    if (lane == 0) red[wid] = mx;
    __syncthreads();
    if (tid == 0) { float m = red[0]; for (int w = 1; w < 8; ++w) m = fmaxf(m, red[w]); red[0] = m; }
    __syncthreads();
    mx = red[0];
    __syncthreads();

    float s = 0.0f;
    for (int j = tid; j < HWSZ; j += blockDim.x) {
        float v = row[j] + ((q_idx[j] == ai) ? 0.0f : -1.0e6f);
        float e = __expf(v - mx);
        Ptf[(size_t)j * HWSZ + h] = e;
        s += e;
    }
    for (int off = 16; off; off >>= 1) s += __shfl_xor(s, off, 32);
    if (lane == 0) red[wid] = s;
    __syncthreads();
    if (tid == 0) { float t = 0.0f; for (int w = 0; w < 8; ++w) t += red[w]; red[0] = t; }
    __syncthreads();
    float inv = 1.0f / red[0];
    for (int j = tid; j < HWSZ; j += blockDim.x)
        Pth[(size_t)j * HWSZ + h] = (_Float16)(Ptf[(size_t)j * HWSZ + h] * inv);
}

// ---------------------------------------------------------------------------
// ConvTranspose2d weights (in, out, 2, 2) fp32 -> f16 Wu[parity][out][in]
// ---------------------------------------------------------------------------
__global__ void prep_wu_k(const float* __restrict__ W0, const float* __restrict__ W1,
                          _Float16* __restrict__ Wu0, _Float16* __restrict__ Wu1) {
    int idx = blockIdx.x * blockDim.x + threadIdx.x;
    const int total = 4 * CDIM * CDIM;
    if (idx >= total) return;
    int p = idx / (CDIM * CDIM);
    int rem = idx - p * CDIM * CDIM;
    int o = rem / CDIM;
    int c = rem - o * CDIM;
    size_t src = ((size_t)c * CDIM + o) * 4 + p;
    Wu0[idx] = (_Float16)W0[src];
    Wu1[idx] = (_Float16)W1[src];
}

// ---------------------------------------------------------------------------
// 3x3 conv weights (Cout, Cin, 3, 3) fp32 -> f16 reordered (Cout, tap, Cin)
// ---------------------------------------------------------------------------
__global__ void prep_cw_k(const float* __restrict__ W, _Float16* __restrict__ Wt,
                          int Cout, int Cin) {
    int idx = blockIdx.x * blockDim.x + threadIdx.x;
    int total = Cout * Cin * 9;
    if (idx >= total) return;
    int o = idx / (Cin * 9);
    int rem = idx - o * Cin * 9;
    int t = rem / Cin;
    int c = rem - t * Cin;
    Wt[idx] = (_Float16)W[((size_t)o * Cin + c) * 9 + t];
}

// ---------------------------------------------------------------------------
// Repack dec (raw .view as (512,1024)) into pixel-major decT[pix][c]
// ---------------------------------------------------------------------------
__global__ void tr_dec_k(const _Float16* __restrict__ s, _Float16* __restrict__ d) {
    int idx = blockIdx.x * blockDim.x + threadIdx.x;
    if (idx >= CDIM * HWSZ) return;
    int cc  = idx >> 10;
    int pix = idx & 1023;
    d[(size_t)pix * CDIM + cc] = s[idx];
}

// ---------------------------------------------------------------------------
// ConvTranspose2d(k=2,s=2) as 4 parity GEMMs, register-blocked 2x4.
// XT: (R*R, 512) pixel-major f16; YT: (2R*2R, 512) pixel-major f16.
// ---------------------------------------------------------------------------
template<int R, int MB, int NB>
__global__ void convt_k(const _Float16* __restrict__ Wu, const float* __restrict__ bias,
                        const _Float16* __restrict__ XT, _Float16* __restrict__ YT) {
    const int N = R * R;
    int grp = blockIdx.x * (blockDim.x >> 5) + (threadIdx.x >> 5);
    int tn0 = grp * NB;
    if (tn0 >= (N >> 4)) return;
    int tm0 = blockIdx.y * MB;
    int p   = blockIdx.z;
    const _Float16* A = Wu + (size_t)p * CDIM * CDIM;
    int lane = threadIdx.x & 31;

    v8f zero = {};
    v8f acc[MB][NB];
#pragma unroll
    for (int im = 0; im < MB; ++im)
#pragma unroll
        for (int in = 0; in < NB; ++in) acc[im][in] = zero;

    for (int kb = 0; kb < CDIM; kb += 32) {
        v16h a[MB], b[NB];
#pragma unroll
        for (int im = 0; im < MB; ++im) a[im] = la_row<false>(A, CDIM, CDIM, tm0 + im, kb, lane);
#pragma unroll
        for (int in = 0; in < NB; ++in) b[in] = lb_wt<false>(XT, CDIM, N, tn0 + in, kb, lane);
#pragma unroll
        for (int im = 0; im < MB; ++im)
#pragma unroll
            for (int in = 0; in < NB; ++in)
                acc[im][in] = __builtin_amdgcn_wmma_f32_16x16x32_f16(
                    false, a[im], false, b[in], (short)0, acc[im][in], false, false);
    }

    int g = lane >> 4;
    const int R2 = 2 * R;
#pragma unroll
    for (int im = 0; im < MB; ++im) {
#pragma unroll
        for (int in = 0; in < NB; ++in) {
            int n = (tn0 + in) * 16 + (lane & 15);
            int yy = 2 * (n / R) + (p >> 1);       // R constexpr power of two
            int xx = 2 * (n & (R - 1)) + (p & 1);
#pragma unroll
            for (int r = 0; r < 8; ++r) {
                int m = (tm0 + im) * 16 + r + (g << 3);
                YT[((size_t)yy * R2 + xx) * CDIM + m] = (_Float16)(acc[im][in][r] + bias[m]);
            }
        }
    }
}

// ---------------------------------------------------------------------------
// 3x3 conv (pad 1, no bias) implicit GEMM with K ordered (tap, channel),
// register-blocked: each B chunk = one neighbor pixel's contiguous channel
// slice of the pixel-major input -> vector loads + chunk-uniform selects.
// ---------------------------------------------------------------------------
template<bool RELU, bool GM, bool OUTF32, int CIN_LOG2, int MB, int NB>
__global__ void conv3_k(const _Float16* __restrict__ XT, const _Float16* __restrict__ W,
                        float* __restrict__ Yf, _Float16* __restrict__ YT, int Cout) {
    const int N = IMR * IMR;
    const int Cin = 1 << CIN_LOG2;
    const int K = Cin * 9;
    int grp = blockIdx.x * (blockDim.x >> 5) + (threadIdx.x >> 5);
    int tn0 = grp * NB;
    if (tn0 >= (N >> 4)) return;
    int tm0 = blockIdx.y * MB;
    int lane = threadIdx.x & 31;
    int g = lane >> 4;

    int y0[NB], x0[NB];
#pragma unroll
    for (int in = 0; in < NB; ++in) {
        int n = (tn0 + in) * 16 + (lane & 15);
        y0[in] = n >> 7;
        x0[in] = n & 127;
    }

    v8f zero = {};
    v8f acc[MB][NB];
#pragma unroll
    for (int im = 0; im < MB; ++im)
#pragma unroll
        for (int in = 0; in < NB; ++in) acc[im][in] = zero;

    for (int kb = 0; kb < K; kb += 32) {
        v16h a[MB], b[NB];
#pragma unroll
        for (int im = 0; im < MB; ++im) a[im] = la_row<GM>(W, K, Cout, tm0 + im, kb, lane);
        int k0 = kb + (g << 4);
        int t  = k0 >> CIN_LOG2;
        int c0 = k0 & (Cin - 1);
        int dy = t / 3, dx = t - dy * 3;   // t < 9: cheap
#pragma unroll
        for (int in = 0; in < NB; ++in) {
            int yy = y0[in] + dy - 1, xx = x0[in] + dx - 1;
            bool inb = (yy >= 0) & (yy < IMR) & (xx >= 0) & (xx < IMR);
            int yc = imin(imax(yy, 0), IMR - 1);
            int xc = imin(imax(xx, 0), IMR - 1);
            const _Float16* p = XT + ((size_t)yc * IMR + xc) * Cin + c0;
            v8h z = {};
            v8h lo = *(const v8h*)p;
            v8h hi = *(const v8h*)(p + 8);
            lo = inb ? lo : z;
            hi = inb ? hi : z;
            b[in] = cat8(lo, hi);
        }
#pragma unroll
        for (int im = 0; im < MB; ++im)
#pragma unroll
            for (int in = 0; in < NB; ++in)
                acc[im][in] = __builtin_amdgcn_wmma_f32_16x16x32_f16(
                    false, a[im], false, b[in], (short)0, acc[im][in], false, false);
    }

#pragma unroll
    for (int im = 0; im < MB; ++im) {
#pragma unroll
        for (int in = 0; in < NB; ++in) {
            int n = (tn0 + in) * 16 + (lane & 15);
#pragma unroll
            for (int r = 0; r < 8; ++r) {
                int m = (tm0 + im) * 16 + r + (g << 3);
                if (!GM || m < Cout) {
                    float v = acc[im][in][r];
                    if (RELU) v = fmaxf(v, 0.0f);
                    if (OUTF32) Yf[(size_t)m * N + n] = v;              // NCHW fp32
                    else        YT[(size_t)n * Cout + m] = (_Float16)v; // pixel-major
                }
            }
        }
    }
}

// ---------------------------------------------------------------------------
// Host orchestration
// ---------------------------------------------------------------------------
extern "C" void kernel_launch(void* const* d_in, const int* in_sizes, int n_in,
                              void* d_out, int out_size, void* d_ws, size_t ws_size,
                              hipStream_t stream) {
    const float* tok   = (const float*)d_in[0];
    const float* supp  = (const float*)d_in[1];
    const float* qry   = (const float*)d_in[2];
    const float* qa_W  = (const float*)d_in[3];
    const float* qa_b  = (const float*)d_in[4];
    const float* ks_W  = (const float*)d_in[5];
    const float* ks_b  = (const float*)d_in[6];
    const float* ka_W  = (const float*)d_in[7];
    const float* ka_b  = (const float*)d_in[8];
    const float* vs_W  = (const float*)d_in[9];
    const float* vs_b  = (const float*)d_in[10];
    const float* W1    = (const float*)d_in[11];
    const float* b1    = (const float*)d_in[12];
    const float* W2    = (const float*)d_in[13];
    const float* b2    = (const float*)d_in[14];
    const float* up0W  = (const float*)d_in[15];
    const float* up0b  = (const float*)d_in[16];
    const float* up1W  = (const float*)d_in[17];
    const float* up1b  = (const float*)d_in[18];
    const float* c3W   = (const float*)d_in[19];
    const float* c1W   = (const float*)d_in[20];
    float* out = (float*)d_out;

    char* wsb = (char*)d_ws;
    size_t off = 0;
    auto alloc = [&](size_t bytes) {
        void* p = wsb + off;
        off += (bytes + 255) & ~(size_t)255;
        return p;
    };
    auto allocf = [&](size_t n) { return (float*)alloc(n * 4); };
    auto alloch = [&](size_t n) { return (_Float16*)alloc(n * 2); };

    // fp32 intermediates
    float* Sas  = allocf((size_t)NAG * HWSZ);
    float* Sqa  = allocf((size_t)HWSZ * NAG);
    float* Lb   = allocf((size_t)HWSZ * HWSZ);
    float* Ptf  = allocf((size_t)HWSZ * HWSZ);
    int*   aidx = (int*)alloc(HWSZ * 4);
    int*   qidx = (int*)alloc(HWSZ * 4);

    // f16 inputs / weights
    _Float16* tokh  = alloch((size_t)BSZ * NAG * CDIM);
    _Float16* supph = alloch((size_t)BSZ * HWSZ * CDIM);
    _Float16* qryh  = alloch((size_t)BSZ * HWSZ * CDIM);
    _Float16* qaWh  = alloch((size_t)CDIM * CDIM);
    _Float16* ksWh  = alloch((size_t)CDIM * CDIM);
    _Float16* kaWh  = alloch((size_t)CDIM * CDIM);
    _Float16* vsWh  = alloch((size_t)CDIM * CDIM);
    _Float16* W1h   = alloch((size_t)DFFD * CDIM);
    _Float16* W2h   = alloch((size_t)CDIM * DFFD);
    _Float16* c3Wh  = alloch((size_t)64 * CDIM * 9);   // reordered (o, tap, c)
    _Float16* c1Wh  = alloch((size_t)3 * 64 * 9);      // reordered (o, tap, c)
    _Float16* Wu0h  = alloch((size_t)4 * CDIM * CDIM);
    _Float16* Wu1h  = alloch((size_t)4 * CDIM * CDIM);

    // f16 per-batch intermediates
    _Float16* qah   = alloch((size_t)NAG * CDIM);
    _Float16* kah   = alloch((size_t)NAG * CDIM);
    _Float16* ksh   = alloch((size_t)HWSZ * CDIM);
    _Float16* qqh   = alloch((size_t)HWSZ * CDIM);
    _Float16* vshT  = alloch((size_t)CDIM * HWSZ);     // transposed (c, h)
    _Float16* Sash  = alloch((size_t)NAG * HWSZ);
    _Float16* Sqah  = alloch((size_t)HWSZ * NAG);
    _Float16* Pth   = alloch((size_t)HWSZ * HWSZ);
    _Float16* attnh = alloch((size_t)HWSZ * CDIM);
    _Float16* Hbh   = alloch((size_t)HWSZ * DFFD);
    _Float16* dech  = alloch((size_t)HWSZ * CDIM);
    _Float16* decT  = alloch((size_t)HWSZ * CDIM);     // pixel-major (1024, 512)
    _Float16* x1T   = alloch((size_t)64 * 64 * CDIM);  // pixel-major
    _Float16* x2T   = alloch((size_t)IMR * IMR * CDIM);// pixel-major
    _Float16* c3T   = alloch((size_t)IMR * IMR * 64);  // pixel-major

    dim3 blk(256);
    // grid helpers: x = ceil((tilesN/NB)/8 waves), y = tilesM/MB
    auto gxw = [](int N, int NB) {
        int tiles = (N + 15) / 16;
        int groups = (tiles + NB - 1) / NB;
        return (unsigned)((groups + 7) / 8);
    };
    auto gyw = [](int M, int MB) {
        int tiles = (M + 15) / 16;
        return (unsigned)((tiles + MB - 1) / MB);
    };
    auto cvt = [&](const float* s, _Float16* d, int n) {
        cvt_h_k<<<dim3((n + 255) / 256), blk, 0, stream>>>(s, d, n);
    };
    const float inv_dk = 0.125f;  // 1/sqrt(512/8)

    // one-time conversions / repacks
    cvt(tok,  tokh,  BSZ * NAG * CDIM);
    cvt(supp, supph, BSZ * HWSZ * CDIM);
    cvt(qry,  qryh,  BSZ * HWSZ * CDIM);
    cvt(qa_W, qaWh,  CDIM * CDIM);
    cvt(ks_W, ksWh,  CDIM * CDIM);
    cvt(ka_W, kaWh,  CDIM * CDIM);
    cvt(vs_W, vsWh,  CDIM * CDIM);
    cvt(W1,   W1h,   DFFD * CDIM);
    cvt(W2,   W2h,   CDIM * DFFD);
    prep_cw_k<<<dim3((64 * CDIM * 9 + 255) / 256), blk, 0, stream>>>(c3W, c3Wh, 64, CDIM);
    prep_cw_k<<<dim3((3 * 64 * 9 + 255) / 256),    blk, 0, stream>>>(c1W, c1Wh, 3, 64);
    prep_wu_k<<<dim3((4 * CDIM * CDIM + 255) / 256), blk, 0, stream>>>(up0W, up1W, Wu0h, Wu1h);

    for (int b = 0; b < BSZ; ++b) {
        const _Float16* tb = tokh  + (size_t)b * NAG  * CDIM;
        const _Float16* sb = supph + (size_t)b * HWSZ * CDIM;
        const _Float16* qb = qryh  + (size_t)b * HWSZ * CDIM;

        // small projections (M=8): 1x1 path with M clamp
        gemm_k<0,0,true, false,false,true, 1,1,1><<<dim3(gxw(CDIM,1), gyw(NAG,1)),  blk, 0, stream>>>(tb, CDIM, qaWh, CDIM, qa_b, nullptr, qah,  CDIM, 0, NAG,  CDIM, CDIM, 1.0f);
        gemm_k<0,0,true, false,false,true, 1,1,1><<<dim3(gxw(CDIM,1), gyw(NAG,1)),  blk, 0, stream>>>(tb, CDIM, kaWh, CDIM, ka_b, nullptr, kah,  CDIM, 0, NAG,  CDIM, CDIM, 1.0f);
        // large projections: 2x4 blocked
        gemm_k<0,0,false,false,false,true, 1,2,4><<<dim3(gxw(CDIM,4), gyw(HWSZ,2)), blk, 0, stream>>>(sb, CDIM, ksWh, CDIM, ks_b, nullptr, ksh,  CDIM, 0, HWSZ, CDIM, CDIM, 1.0f);
        gemm_k<0,0,false,false,false,true, 1,2,4><<<dim3(gxw(CDIM,4), gyw(HWSZ,2)), blk, 0, stream>>>(qb, CDIM, qaWh, CDIM, qa_b, nullptr, qqh,  CDIM, 0, HWSZ, CDIM, CDIM, 1.0f); // qq uses qa_linear
        gemm_k<0,0,false,false,false,true, 3,2,4><<<dim3(gxw(CDIM,4), gyw(HWSZ,2)), blk, 0, stream>>>(sb, CDIM, vsWh, CDIM, vs_b, nullptr, vshT, 0, HWSZ, HWSZ, CDIM, CDIM, 1.0f); // vs stored transposed (c,h)

        // scores
        gemm_k<0,0,true, false,false,false,2,1,4><<<dim3(gxw(HWSZ,4), gyw(NAG,1)),  blk, 0, stream>>>(qah, CDIM, ksh, CDIM, nullptr, Sas, Sash, HWSZ, 0, NAG,  HWSZ, CDIM, inv_dk);
        gemm_k<0,0,false,true, false,false,2,2,1><<<dim3(gxw(NAG,1),  gyw(HWSZ,2)), blk, 0, stream>>>(qqh, CDIM, kah, CDIM, nullptr, Sqa, Sqah, NAG,  0, HWSZ, NAG,  CDIM, inv_dk);

        argmax_k<<<dim3(4), blk, 0, stream>>>(Sas, Sqa, aidx, qidx);

        // L[h,j] = sum_a Sas[a,h] * Sqa[j,a]  (K=8 zero-guarded, 2x4 blocked)
        gemm_k<1,2,false,false,false,false,0,2,4><<<dim3(gxw(HWSZ,4), gyw(HWSZ,2)), blk, 0, stream>>>(Sash, HWSZ, Sqah, NAG, nullptr, Lb, nullptr, HWSZ, 0, HWSZ, HWSZ, NAG, 1.0f);

        softmax_t_k<<<dim3(HWSZ), blk, 0, stream>>>(Lb, Ptf, Pth, aidx, qidx);

        // attn[i,c] = sum_h Pt[i,h] * vsT[c,h]
        gemm_k<0,0,false,false,false,false,1,2,4><<<dim3(gxw(CDIM,4), gyw(HWSZ,2)), blk, 0, stream>>>(Pth, HWSZ, vshT, HWSZ, nullptr, nullptr, attnh, CDIM, 0, HWSZ, CDIM, HWSZ, 1.0f);

        // FFN
        gemm_k<0,0,false,false,true, true, 1,2,4><<<dim3(gxw(DFFD,4), gyw(HWSZ,2)), blk, 0, stream>>>(attnh, CDIM, W1h, CDIM, b1, nullptr, Hbh,  DFFD, 0, HWSZ, DFFD, CDIM, 1.0f);
        gemm_k<0,0,false,false,false,true, 1,2,4><<<dim3(gxw(CDIM,4), gyw(HWSZ,2)), blk, 0, stream>>>(Hbh,  DFFD, W2h, DFFD, b2, nullptr, dech, CDIM, 0, HWSZ, CDIM, DFFD, 1.0f);

        // raw .view to (512,32,32) then repack pixel-major
        tr_dec_k<<<dim3((CDIM * HWSZ + 255) / 256), blk, 0, stream>>>(dech, decT);

        convt_k<32,2,4><<<dim3(gxw(32 * 32, 4), gyw(CDIM, 2), 4), blk, 0, stream>>>(Wu0h, up0b, decT, x1T);
        convt_k<64,2,4><<<dim3(gxw(64 * 64, 4), gyw(CDIM, 2), 4), blk, 0, stream>>>(Wu1h, up1b, x1T,  x2T);

        conv3_k<true, false,false,9,2,2><<<dim3(gxw(IMR * IMR, 2), gyw(64, 2)), blk, 0, stream>>>(x2T, c3Wh, nullptr, c3T, 64);
        conv3_k<false,true, true, 6,1,1><<<dim3(gxw(IMR * IMR, 1), gyw(3, 1)),  blk, 0, stream>>>(c3T, c1Wh, out + (size_t)b * 3 * IMR * IMR, nullptr, 3);
    }

    (void)in_sizes; (void)n_in; (void)out_size; (void)ws_size;
}
